// Transformer_13804024890058
// MI455X (gfx1250) — compile-verified, matrix operand order checked by reading, not run on hardware
//
#include <hip/hip_runtime.h>

typedef unsigned short u16;
typedef __attribute__((ext_vector_type(16))) __bf16 v16bf;
typedef __attribute__((ext_vector_type(8)))  float  v8f;
typedef unsigned int u32x4 __attribute__((ext_vector_type(4)));
typedef int          i32x8 __attribute__((ext_vector_type(8)));
typedef int          i32x4 __attribute__((ext_vector_type(4)));

#if defined(__AMDGCN__) && __has_builtin(__builtin_amdgcn_tensor_load_to_lds)
#define USE_TDM 1
#else
#define USE_TDM 0
#endif

#define WMMA_BF16(a, b, c) \
  __builtin_amdgcn_wmma_f32_16x16x32_bf16(false, (a), false, (b), (short)0, (c), false, false)

__device__ __forceinline__ u16 f2bf(float x) {
  union { float f; unsigned u; } a; a.f = x;
  unsigned r = a.u + 0x7FFFu + ((a.u >> 16) & 1u);
  return (u16)(r >> 16);
}

__device__ __forceinline__ v8f zero8() {
  v8f z = {0.f, 0.f, 0.f, 0.f, 0.f, 0.f, 0.f, 0.f};
  return z;
}

// Load a 16x32 bf16 A/B fragment. p must point at (row's) K-base (16B aligned).
// Layout (CDNA5 16-bit A/B): lane&15 selects row; lanes>=16 take K chunks +8;
// VGPRs 0-3 = K[8*hi .. 8*hi+8), VGPRs 4-7 = K[16+8*hi .. +8).
__device__ __forceinline__ v16bf frag_ld(const u16* p, int lhi) {
  const u16* q = p + 8 * lhi;
  union { uint4 u[2]; v16bf f; } uu;
  uu.u[0] = *(const uint4*)(q);
  uu.u[1] = *(const uint4*)(q + 16);
  return uu.f;
}

template <int N>
__device__ __forceinline__ void wait_tensorcnt() {
#if defined(__AMDGCN__) && __has_builtin(__builtin_amdgcn_s_wait_tensorcnt)
  __builtin_amdgcn_s_wait_tensorcnt((short)N);
#elif defined(__AMDGCN__)
  asm volatile("s_wait_tensorcnt %0" ::"i"(N) : "memory");
#endif
}

#if USE_TDM
// TDM: DMA one 128-row x 32-col bf16 tile (row stride = row_len elems) from
// global into LDS at byte offset lds_off, padding 4 DWORDs after every
// 16 DWORDs stored -> LDS row stride 40 halves (matches frag_ld layout).
__device__ __forceinline__ void tdm_tile_load(const u16* gsrc, unsigned lds_off,
                                              int row_len) {
  unsigned long long ga = (unsigned long long)(uintptr_t)gsrc;
  u32x4 g0;
  g0[0] = 1u;                                            // count=1 (valid), no gather
  g0[1] = lds_off;                                       // lds_addr (bytes)
  g0[2] = (unsigned)ga;                                  // global_addr[31:0]
  g0[3] = (unsigned)((ga >> 32) & 0x01FFFFFFu) | (2u << 30); // addr[56:32], type=2
  i32x8 g1;
  g1[0] = (1 << 16)      // data_size = 1 -> 2 bytes
        | (1 << 20)      // pad_enable
        | (3 << 22)      // pad_interval: 16 DWORDs
        | (3 << 25);     // pad_amount: 4 DWORDs
  g1[1] = (32 << 16);            // tensor_dim0 = 32 (bits 79:48, low part)
  g1[2] = (128 << 16);           // tensor_dim0 hi=0 | tensor_dim1 = 128 (bits 111:80)
  g1[3] = (32 << 16);            // tensor_dim1 hi=0 | tile_dim0 = 32 (bits 127:112)
  g1[4] = 128;                   // tile_dim1 = 128, tile_dim2 = 0
  g1[5] = row_len;               // tensor_dim0_stride (elems), bits 207:160 low
  g1[6] = 0;
  g1[7] = 0;
  i32x4 gz = {0, 0, 0, 0};
#if __clang_major__ >= 23
  i32x8 gz8 = {0, 0, 0, 0, 0, 0, 0, 0};
  __builtin_amdgcn_tensor_load_to_lds(g0, g1, gz, gz, gz8, 0);
#else
  __builtin_amdgcn_tensor_load_to_lds(g0, g1, gz, gz, 0);
#endif
}
#endif

// ---------------------------------------------------------------------------
// GEMM: C[Nr,Nc] = A[Nr,K](bf16) * W[Nc,K](bf16)^T + bias
// block tile 128x128, 8 waves (4 in M x 2 in N), wave tile 32x64, K-step 32.
// Tile staging via TDM (double buffered) when available.
// ---------------------------------------------------------------------------
__global__ __launch_bounds__(256) void gemm_bf16(
    const u16* __restrict__ A, const u16* __restrict__ W,
    const float* __restrict__ bias, float* __restrict__ C,
    int K, int Nc)
{
  const int tid  = threadIdx.x;
  const int wave = tid >> 5, lane = tid & 31;
  const int lrow = lane & 15, lhi = lane >> 4;
  const int bm = blockIdx.y * 128, bn = blockIdx.x * 128;
  const int wm = (wave >> 1) * 32, wn = (wave & 1) * 64;

  v8f acc[2][4];
#pragma unroll
  for (int t = 0; t < 2; ++t)
#pragma unroll
    for (int u = 0; u < 4; ++u) acc[t][u] = zero8();

#if USE_TDM
  __shared__ u16 sA[2][128 * 40];
  __shared__ u16 sW[2][128 * 40];
  const unsigned ldsA0 = (unsigned)(uintptr_t)&sA[0][0];
  const unsigned ldsA1 = (unsigned)(uintptr_t)&sA[1][0];
  const unsigned ldsW0 = (unsigned)(uintptr_t)&sW[0][0];
  const unsigned ldsW1 = (unsigned)(uintptr_t)&sW[1][0];
  const u16* gAt = A + (size_t)bm * K;
  const u16* gWt = W + (size_t)bn * K;
  const int nk = K >> 5;

  if (wave == 0) {
    tdm_tile_load(gAt, ldsA0, K);
    tdm_tile_load(gWt, ldsW0, K);
  }
  for (int kk = 0; kk < nk; ++kk) {
    const int cur = kk & 1;
    if (wave == 0) {
      if (kk + 1 < nk) {
        tdm_tile_load(gAt + (size_t)(kk + 1) * 32, cur ? ldsA0 : ldsA1, K);
        tdm_tile_load(gWt + (size_t)(kk + 1) * 32, cur ? ldsW0 : ldsW1, K);
        wait_tensorcnt<2>();   // tile kk's two DMAs complete (in-order)
      } else {
        wait_tensorcnt<0>();
      }
    }
    __syncthreads();
    v16bf af[2], wf[4];
#pragma unroll
    for (int t = 0; t < 2; ++t)
      af[t] = frag_ld(sA[cur] + (wm + t * 16 + lrow) * 40, lhi);
#pragma unroll
    for (int u = 0; u < 4; ++u)
      wf[u] = frag_ld(sW[cur] + (wn + u * 16 + lrow) * 40, lhi);
#pragma unroll
    for (int t = 0; t < 2; ++t)
#pragma unroll
      for (int u = 0; u < 4; ++u)
        acc[t][u] = WMMA_BF16(af[t], wf[u], acc[t][u]);
    __syncthreads();           // all reads done before buffer is re-filled
  }
#else
  __shared__ u16 sA1[128 * 40];   // padded stride 40 halves (80B, 16B-aligned)
  __shared__ u16 sW1[128 * 40];
  const int lr = tid >> 1;
  const int lc = (tid & 1) * 16;
  const u16* gA = A + (size_t)(bm + lr) * K + lc;
  const u16* gW = W + (size_t)(bn + lr) * K + lc;
  u16* stA = sA1 + lr * 40 + lc;
  u16* stW = sW1 + lr * 40 + lc;

  for (int k0 = 0; k0 < K; k0 += 32) {
    uint4 a0 = *(const uint4*)(gA + k0);
    uint4 a1 = *(const uint4*)(gA + k0 + 8);
    uint4 w0 = *(const uint4*)(gW + k0);
    uint4 w1 = *(const uint4*)(gW + k0 + 8);
    if (k0 + 32 < K) {
      __builtin_prefetch(gA + k0 + 32, 0, 1);
      __builtin_prefetch(gW + k0 + 32, 0, 1);
    }
    __syncthreads();
    *(uint4*)(stA) = a0; *(uint4*)(stA + 8) = a1;
    *(uint4*)(stW) = w0; *(uint4*)(stW + 8) = w1;
    __syncthreads();

    v16bf af[2], wf[4];
#pragma unroll
    for (int t = 0; t < 2; ++t) af[t] = frag_ld(sA1 + (wm + t * 16 + lrow) * 40, lhi);
#pragma unroll
    for (int u = 0; u < 4; ++u) wf[u] = frag_ld(sW1 + (wn + u * 16 + lrow) * 40, lhi);
#pragma unroll
    for (int t = 0; t < 2; ++t)
#pragma unroll
      for (int u = 0; u < 4; ++u)
        acc[t][u] = WMMA_BF16(af[t], wf[u], acc[t][u]);
  }
#endif

#pragma unroll
  for (int t = 0; t < 2; ++t) {
    const int row0 = bm + wm + t * 16 + lhi * 8;   // + v  (C layout: M = v + 8*hi)
#pragma unroll
    for (int u = 0; u < 4; ++u) {
      const int col = bn + wn + u * 16 + lrow;     // N = lane&15
      const float bv = bias ? bias[col] : 0.f;
      float* cp = C + (size_t)row0 * Nc + col;
#pragma unroll
      for (int v = 0; v < 8; ++v)
        cp[(size_t)v * Nc] = acc[t][u][v] + bv;
    }
  }
}

// ---------------------------------------------------------------------------
// Flash attention: one wave per 16-query tile. q:[BH,Sq,64] k:[BH,Skv,64]
// vt:[BH,64,Skv]  out:[B,Sq,512] bf16. q pre-scaled by 1/sqrt(dh).
// ---------------------------------------------------------------------------
__global__ __launch_bounds__(32) void flash_attn(
    const u16* __restrict__ q, const u16* __restrict__ k, const u16* __restrict__ vt,
    u16* __restrict__ out, int Sq, int Skv, int causal)
{
  __shared__ u16 sP[16 * 32];
  const int lane = threadIdx.x;
  const int lrow = lane & 15, lhi = lane >> 4;
  const int ntile = Sq >> 4;
  const int qt = blockIdx.x % ntile;
  const int bh = blockIdx.x / ntile;
  const int b = bh >> 3, h = bh & 7;               // H = 8
  const int qbase = qt * 16;

  const u16* qp = q  + (size_t)bh * Sq  * 64;
  const u16* kp = k  + (size_t)bh * Skv * 64;
  const u16* vp = vt + (size_t)bh * 64  * Skv;

  v16bf aq0, aq1;
  {
    const u16* r = qp + (size_t)(qbase + lrow) * 64;
    aq0 = frag_ld(r, lhi);        // dh 0..31
    aq1 = frag_ld(r + 32, lhi);   // dh 32..63
  }

  v8f o[4];
  float rmax[8], rsum[8];
#pragma unroll
  for (int f = 0; f < 4; ++f) o[f] = zero8();
#pragma unroll
  for (int v = 0; v < 8; ++v) { rmax[v] = -1e30f; rsum[v] = 0.f; }

  const int jend = causal ? (qbase + 16) : Skv;
  for (int j = 0; j < jend; j += 32) {
    v8f s0 = zero8(), s1 = zero8();
    {
      const u16* kr0 = kp + (size_t)(j + lrow) * 64;
      const u16* kr1 = kp + (size_t)(j + 16 + lrow) * 64;
      s0 = WMMA_BF16(aq0, frag_ld(kr0,      lhi), s0);
      s0 = WMMA_BF16(aq1, frag_ld(kr0 + 32, lhi), s0);
      s1 = WMMA_BF16(aq0, frag_ld(kr1,      lhi), s1);
      s1 = WMMA_BF16(aq1, frag_ld(kr1 + 32, lhi), s1);
    }
    if (causal && (j + 31 > qbase)) {
#pragma unroll
      for (int v = 0; v < 8; ++v) {
        const int qrow = qbase + v + 8 * lhi;
        if (j + lrow      > qrow) s0[v] = -1e30f;
        if (j + 16 + lrow > qrow) s1[v] = -1e30f;
      }
    }
    float scl[8];
#pragma unroll
    for (int v = 0; v < 8; ++v) {
      float bmax = fmaxf(s0[v], s1[v]);
      for (int m = 8; m; m >>= 1) bmax = fmaxf(bmax, __shfl_xor(bmax, m, 32));
      const float mn = fmaxf(rmax[v], bmax);
      const float sc = __expf(rmax[v] - mn);
      rmax[v] = mn;
      const float p0 = __expf(s0[v] - mn);
      const float p1 = __expf(s1[v] - mn);
      s0[v] = p0; s1[v] = p1;
      float rs = p0 + p1;
      for (int m = 8; m; m >>= 1) rs += __shfl_xor(rs, m, 32);
      rsum[v] = rsum[v] * sc + rs;
      scl[v] = sc;
    }
#pragma unroll
    for (int f = 0; f < 4; ++f)
#pragma unroll
      for (int v = 0; v < 8; ++v) o[f][v] *= scl[v];

    // P: C-layout -> A-layout via LDS bounce (same-wave LDS is in-order)
#pragma unroll
    for (int v = 0; v < 8; ++v) {
      const int pr = v + 8 * lhi;
      sP[pr * 32 + lrow]      = f2bf(s0[v]);
      sP[pr * 32 + 16 + lrow] = f2bf(s1[v]);
    }
    const v16bf pf = frag_ld(sP + lrow * 32, lhi);
#pragma unroll
    for (int f = 0; f < 4; ++f) {
      const u16* vr = vp + (size_t)(16 * f + lrow) * Skv + j;
      o[f] = WMMA_BF16(pf, frag_ld(vr, lhi), o[f]);
    }
  }
#pragma unroll
  for (int f = 0; f < 4; ++f)
#pragma unroll
    for (int v = 0; v < 8; ++v) {
      const int row = qbase + v + 8 * lhi;
      const int col = h * 64 + 16 * f + lrow;
      out[((size_t)b * Sq + row) * 512 + col] = f2bf(o[f][v] / rsum[v]);
    }
}

// ---------------------------------------------------------------------------
// Elementwise helpers
// ---------------------------------------------------------------------------
__global__ void k_f32_to_bf16(const float* __restrict__ in, u16* __restrict__ out, size_t n) {
  size_t i = (size_t)blockIdx.x * blockDim.x + threadIdx.x;
  if (i < n) out[i] = f2bf(in[i]);
}

__global__ void k_embed(const int* __restrict__ seq, const float* __restrict__ emb,
                        const float* __restrict__ pos, float* __restrict__ xf,
                        u16* __restrict__ xb) {
  size_t i = (size_t)blockIdx.x * blockDim.x + threadIdx.x;   // over 4096*512
  if (i >= (size_t)4096 * 512) return;
  const int e = (int)(i & 511);
  const int t = (int)(i >> 9);
  const int s = t & 2047;                                      // S = 2048
  const float v = emb[(size_t)seq[t] * 512 + e] + pos[(size_t)s * 512 + e];
  xf[i] = v; xb[i] = f2bf(v);
}

__global__ void k_scatter_qkv(const float* __restrict__ in, u16* __restrict__ qs,
                              u16* __restrict__ ks, u16* __restrict__ vts,
                              int S_, size_t total) {
  size_t i = (size_t)blockIdx.x * blockDim.x + threadIdx.x;
  if (i >= total) return;
  const int c = (int)(i % 1536);
  const size_t t = i / 1536;
  const int bb = (int)(t / S_), s = (int)(t % S_);
  const int part = c >> 9, e = c & 511, h = e >> 6, d = e & 63;
  const float v = in[i];
  const size_t bh = (size_t)bb * 8 + h;
  if (part == 0)      qs[(bh * S_ + s) * 64 + d] = f2bf(v * 0.125f);  // 1/sqrt(64)
  else if (part == 1) ks[(bh * S_ + s) * 64 + d] = f2bf(v);
  else                vts[(bh * 64 + d) * S_ + s] = f2bf(v);
}

__global__ void k_scatter_q(const float* __restrict__ in, u16* __restrict__ qx,
                            int S_, size_t total) {
  size_t i = (size_t)blockIdx.x * blockDim.x + threadIdx.x;
  if (i >= total) return;
  const int e = (int)(i & 511);
  const size_t t = i >> 9;
  const int bb = (int)(t / S_), s = (int)(t % S_);
  const int h = e >> 6, d = e & 63;
  qx[(((size_t)bb * 8 + h) * S_ + s) * 64 + d] = f2bf(in[i] * 0.125f);
}

__global__ void k_scatter_kv(const float* __restrict__ in, u16* __restrict__ kx,
                             u16* __restrict__ vtx, int M_, size_t total) {
  size_t i = (size_t)blockIdx.x * blockDim.x + threadIdx.x;
  if (i >= total) return;
  const int c = (int)(i & 1023);
  const size_t t = i >> 10;
  const int bb = (int)(t / M_), m = (int)(t % M_);
  const int part = c >> 9, e = c & 511, h = e >> 6, d = e & 63;
  const float v = in[i];
  const size_t bh = (size_t)bb * 8 + h;
  if (part == 0) kx[(bh * M_ + m) * 64 + d] = f2bf(v);
  else           vtx[(bh * 64 + d) * M_ + m] = f2bf(v);
}

__global__ void k_gelu(const float* __restrict__ in, u16* __restrict__ out, size_t n) {
  size_t i = (size_t)blockIdx.x * blockDim.x + threadIdx.x;
  if (i >= n) return;
  const float x = in[i];
  out[i] = f2bf(0.5f * x * (1.f + erff(x * 0.70710678118f)));
}

// y = LN(x + d) * g + b   (E = 512 hard-coded; one block of 256 per row)
__global__ __launch_bounds__(256) void k_add_ln(
    const float* __restrict__ x, const float* __restrict__ d,
    const float* __restrict__ g, const float* __restrict__ bt,
    float* __restrict__ yf, u16* __restrict__ yb)
{
  const int row = blockIdx.x, t = threadIdx.x;
  const size_t base = (size_t)row * 512;
  float v0 = x[base + t];
  float v1 = x[base + t + 256];
  if (d) { v0 += d[base + t]; v1 += d[base + t + 256]; }
  float sum = v0 + v1, ssq = v0 * v0 + v1 * v1;
  for (int m = 16; m; m >>= 1) {
    sum += __shfl_xor(sum, m, 32);
    ssq += __shfl_xor(ssq, m, 32);
  }
  __shared__ float s1a[8], s2a[8];
  const int w = t >> 5;
  if ((t & 31) == 0) { s1a[w] = sum; s2a[w] = ssq; }
  __syncthreads();
  float S = 0.f, Q = 0.f;
#pragma unroll
  for (int i = 0; i < 8; ++i) { S += s1a[i]; Q += s2a[i]; }
  const float mean = S * (1.f / 512.f);
  const float var  = Q * (1.f / 512.f) - mean * mean;
  const float rs = rsqrtf(var + 1e-5f);
  const float o0 = (v0 - mean) * rs * g[t] + bt[t];
  const float o1 = (v1 - mean) * rs * g[t + 256] + bt[t + 256];
  yf[base + t] = o0;        yb[base + t] = f2bf(o0);
  yf[base + t + 256] = o1;  yb[base + t + 256] = f2bf(o1);
}

// ---------------------------------------------------------------------------
// Host orchestration
// ---------------------------------------------------------------------------
extern "C" void kernel_launch(void* const* d_in, const int* in_sizes, int n_in,
                              void* d_out, int out_size, void* d_ws, size_t ws_size,
                              hipStream_t stream) {
  (void)in_sizes; (void)n_in; (void)out_size; (void)ws_size;
  const float* encoded     = (const float*)d_in[0];
  const int*   seq         = (const int*)  d_in[1];
  const float* input_embed = (const float*)d_in[2];
  const float* pos_embed   = (const float*)d_in[3];
  const float* output_bias = (const float*)d_in[4];
  const float* self_in_w   = (const float*)d_in[5];
  const float* self_in_b   = (const float*)d_in[6];
  const float* self_out_w  = (const float*)d_in[7];
  const float* self_out_b  = (const float*)d_in[8];
  const float* cross_in_w  = (const float*)d_in[9];
  const float* cross_in_b  = (const float*)d_in[10];
  const float* cross_out_w = (const float*)d_in[11];
  const float* cross_out_b = (const float*)d_in[12];
  const float* self_ng     = (const float*)d_in[13];
  const float* self_nb     = (const float*)d_in[14];
  const float* cross_ng    = (const float*)d_in[15];
  const float* cross_nb    = (const float*)d_in[16];
  const float* mlp_ng      = (const float*)d_in[17];
  const float* mlp_nb      = (const float*)d_in[18];
  const float* lin1_w      = (const float*)d_in[19];
  const float* lin1_b      = (const float*)d_in[20];
  const float* lin2_w      = (const float*)d_in[21];
  const float* lin2_b      = (const float*)d_in[22];
  const float* final_g     = (const float*)d_in[23];
  const float* final_b     = (const float*)d_in[24];

  const int S = 2048, M = 1024, T = 4096, Tm = 2048, L = 6;

  char* wsp = (char*)d_ws;
  size_t off = 0;
  auto alloc = [&](size_t bytes) -> char* {
    char* p = wsp + off;
    off = (off + bytes + 255) & ~(size_t)255;
    return p;
  };

  u16* wSelfIn  = (u16*)alloc((size_t)L * 1536 * 512 * 2);
  u16* wSelfOut = (u16*)alloc((size_t)L * 512 * 512 * 2);
  u16* wCrossIn = (u16*)alloc((size_t)L * 1536 * 512 * 2);
  u16* wCrossOut= (u16*)alloc((size_t)L * 512 * 512 * 2);
  u16* wLin1    = (u16*)alloc((size_t)L * 2048 * 512 * 2);
  u16* wLin2    = (u16*)alloc((size_t)L * 512 * 2048 * 2);
  u16* wEmb     = (u16*)alloc((size_t)8192 * 512 * 2);
  u16* encB     = (u16*)alloc((size_t)Tm * 512 * 2);
  float* xF     = (float*)alloc((size_t)T * 512 * 4);
  u16*   xB     = (u16*)  alloc((size_t)T * 512 * 2);
  float* yF     = (float*)alloc((size_t)T * 512 * 4);
  u16*   yB     = (u16*)  alloc((size_t)T * 512 * 2);
  float* y2F    = (float*)alloc((size_t)T * 512 * 4);
  u16*   y2B    = (u16*)  alloc((size_t)T * 512 * 2);
  float* bigF   = (float*)alloc((size_t)T * 2048 * 4);   // qkv / lin1 out
  float* smallF = (float*)alloc((size_t)T * 512 * 4);    // proj outputs
  float* kvF    = (float*)alloc((size_t)Tm * 1024 * 4);
  u16* qs  = (u16*)alloc((size_t)16 * S * 64 * 2);
  u16* ks  = (u16*)alloc((size_t)16 * S * 64 * 2);
  u16* vts = (u16*)alloc((size_t)16 * 64 * S * 2);
  u16* qx  = (u16*)alloc((size_t)16 * S * 64 * 2);
  u16* kx  = (u16*)alloc((size_t)16 * M * 64 * 2);
  u16* vtx = (u16*)alloc((size_t)16 * 64 * M * 2);
  u16* attnB  = (u16*)alloc((size_t)T * 512 * 2);
  u16* cattnB = (u16*)alloc((size_t)T * 512 * 2);
  u16* hB     = (u16*)alloc((size_t)T * 2048 * 2);

  auto conv = [&](const float* src, u16* dst, size_t n) {
    k_f32_to_bf16<<<dim3((unsigned)((n + 255) / 256)), 256, 0, stream>>>(src, dst, n);
  };
  auto gemm = [&](const u16* A, const u16* W, const float* bias, float* C,
                  int Nr, int Nc, int K) {
    gemm_bf16<<<dim3(Nc / 128, Nr / 128), 256, 0, stream>>>(A, W, bias, C, K, Nc);
  };
  auto attn = [&](const u16* q, const u16* k, const u16* vt, u16* o,
                  int Sq, int Skv, int causal) {
    flash_attn<<<dim3(16 * (Sq / 16)), 32, 0, stream>>>(q, k, vt, o, Sq, Skv, causal);
  };

  // Weight / input conversions to bf16
  conv(self_in_w,  wSelfIn,  (size_t)L * 1536 * 512);
  conv(self_out_w, wSelfOut, (size_t)L * 512 * 512);
  conv(cross_in_w, wCrossIn, (size_t)L * 1536 * 512);
  conv(cross_out_w,wCrossOut,(size_t)L * 512 * 512);
  conv(lin1_w,     wLin1,    (size_t)L * 2048 * 512);
  conv(lin2_w,     wLin2,    (size_t)L * 512 * 2048);
  conv(input_embed,wEmb,     (size_t)8192 * 512);
  conv(encoded,    encB,     (size_t)Tm * 512);

  // Embedding
  k_embed<<<dim3((unsigned)(((size_t)T * 512 + 255) / 256)), 256, 0, stream>>>(
      seq, input_embed, pos_embed, xF, xB);

  for (int l = 0; l < L; ++l) {
    // --- self attention ---
    gemm(xB, wSelfIn + (size_t)l * 1536 * 512, self_in_b + (size_t)l * 1536,
         bigF, T, 1536, 512);
    {
      size_t n = (size_t)T * 1536;
      k_scatter_qkv<<<dim3((unsigned)((n + 255) / 256)), 256, 0, stream>>>(
          bigF, qs, ks, vts, S, n);
    }
    attn(qs, ks, vts, attnB, S, S, 1);
    gemm(attnB, wSelfOut + (size_t)l * 512 * 512, self_out_b + (size_t)l * 512,
         smallF, T, 512, 512);
    k_add_ln<<<dim3(T), 256, 0, stream>>>(xF, smallF,
        self_ng + (size_t)l * 512, self_nb + (size_t)l * 512, yF, yB);

    // --- cross attention ---
    gemm(yB, wCrossIn + (size_t)l * 1536 * 512, cross_in_b + (size_t)l * 1536,
         smallF, T, 512, 512);                                   // q proj
    {
      size_t n = (size_t)T * 512;
      k_scatter_q<<<dim3((unsigned)((n + 255) / 256)), 256, 0, stream>>>(
          smallF, qx, S, n);
    }
    gemm(encB, wCrossIn + ((size_t)l * 1536 + 512) * 512,
         cross_in_b + (size_t)l * 1536 + 512, kvF, Tm, 1024, 512); // k,v proj
    {
      size_t n = (size_t)Tm * 1024;
      k_scatter_kv<<<dim3((unsigned)((n + 255) / 256)), 256, 0, stream>>>(
          kvF, kx, vtx, M, n);
    }
    attn(qx, kx, vtx, cattnB, S, M, 0);
    gemm(cattnB, wCrossOut + (size_t)l * 512 * 512, cross_out_b + (size_t)l * 512,
         smallF, T, 512, 512);
    k_add_ln<<<dim3(T), 256, 0, stream>>>(yF, smallF,
        cross_ng + (size_t)l * 512, cross_nb + (size_t)l * 512, y2F, y2B);

    // --- MLP ---
    gemm(y2B, wLin1 + (size_t)l * 2048 * 512, lin1_b + (size_t)l * 2048,
         bigF, T, 2048, 512);
    {
      size_t n = (size_t)T * 2048;
      k_gelu<<<dim3((unsigned)((n + 255) / 256)), 256, 0, stream>>>(bigF, hB, n);
    }
    gemm(hB, wLin2 + (size_t)l * 512 * 2048, lin2_b + (size_t)l * 512,
         smallF, T, 512, 2048);
    k_add_ln<<<dim3(T), 256, 0, stream>>>(y2F, smallF,
        mlp_ng + (size_t)l * 512, mlp_nb + (size_t)l * 512, xF, xB);
  }

  // Final LN + shared-embedding output projection
  k_add_ln<<<dim3(T), 256, 0, stream>>>(xF, (const float*)nullptr,
                                        final_g, final_b, yF, xB);
  gemm(xB, wEmb, output_bias, (float*)d_out, T, 8192, 512);
}
